// WindowAttention3D_7292854469154
// MI455X (gfx1250) — matrix operand order repaired and checked by reading, not compiled, not run on hardware
//
#include <hip/hip_runtime.h>

typedef unsigned short u16;
typedef unsigned int   u32;
typedef __attribute__((ext_vector_type(16))) __bf16 v16bf;
typedef __attribute__((ext_vector_type(2)))  __bf16 v2bf;
typedef __attribute__((ext_vector_type(8)))  float  v8f;
typedef __attribute__((ext_vector_type(4)))  int    v4i;

union Frag { u32 u[8]; v16bf v; };

// Native f32->bf16 (RNE) — lowers to v_cvt_bf16_f32 on gfx1250.
__device__ __forceinline__ u16 f2bf(float x) {
    union { __bf16 h; u16 s; } c;
    c.h = (__bf16)x;
    return c.s;
}

// Packed pair conversion — one v_cvt_pk_bf16_f32 produces the final dword.
__device__ __forceinline__ u32 f2bf2(float a, float b) {
    union { v2bf v; u32 u; } c;
    c.v.x = (__bf16)a;
    c.v.y = (__bf16)b;
    return c.u;
}

__device__ __forceinline__ v8f wmma_bf16(Frag a, Frag b, v8f c) {
    // (neg_a, A, neg_b, B, c_mod, C, reuse_a, reuse_b)
    return __builtin_amdgcn_wmma_f32_16x16x32_bf16(false, a.v, false, b.v,
                                                   (short)0, c, false, false);
}

// ---- CDNA5 async global->LDS copy (ASYNCcnt), guarded so compile can't fail.
#if __has_builtin(__builtin_amdgcn_global_load_async_to_lds_b128)
#define HAVE_ASYNC_LDS 1
typedef __attribute__((address_space(1))) v4i GV4;   // global int4*
typedef __attribute__((address_space(3))) v4i LV4;   // LDS    int4*
__device__ __forceinline__ void async_cp16(const void* g, void* l) {
    __builtin_amdgcn_global_load_async_to_lds_b128((GV4*)g, (LV4*)l, 0, 0);
}
__device__ __forceinline__ void async_wait0() {
    asm volatile("s_wait_asynccnt 0x0" ::: "memory");
}
#else
#define HAVE_ASYNC_LDS 0
#endif

// ---------------------------------------------------------------------------
// Kernel 0: convert weights fp32 -> bf16
// ---------------------------------------------------------------------------
__global__ __launch_bounds__(256) void convert_w(const float* __restrict__ wq,
                                                 const float* __restrict__ wo,
                                                 u16* __restrict__ wqb,
                                                 u16* __restrict__ wob) {
    int i = blockIdx.x * 256 + threadIdx.x;
    if (i < 196608) {
        wqb[i] = f2bf(wq[i]);
    } else {
        int j = i - 196608;
        if (j < 65536) wob[j] = f2bf(wo[j]);
    }
}

// ---------------------------------------------------------------------------
// Kernel 1: QKV projection with fused cyclic shift.
// M = 110592 tokens in window-major (shifted) order, K = 256, N = 768.
// 64x64 tile / block, 128 threads (4 waves), each wave 32x32.
// Output: Q/K/V bf16 laid out [win][head][pos(64)][d(32)], Q pre-scaled.
// ---------------------------------------------------------------------------
__global__ __launch_bounds__(128) void qkv_gemm(const float* __restrict__ x,
                                                const u16* __restrict__ wqkv,
                                                u16* __restrict__ Qb,
                                                u16* __restrict__ Kb,
                                                u16* __restrict__ Vb) {
    __shared__ __align__(16) u16 As[64 * 32];   // [row][k]
    __shared__ __align__(16) u16 Bt[64 * 32];   // [n][k]  (transposed)

    const int bid  = blockIdx.x;
    const int mT   = bid / 12;          // window index (64 tokens per window)
    const int nT   = bid % 12;          // 64-wide column tile of 768
    const int tid  = threadIdx.x;
    const int lane = tid & 31;
    const int half = lane >> 4;
    const int lm   = lane & 15;
    const int wave = tid >> 5;
    const int wr   = (wave >> 1) << 5;  // wave row offset in 64x64 tile
    const int wc   = (wave & 1) << 5;   // wave col offset

    // A-staging assignment: thread handles row = tid>>1, 16 k's.
    const int arow  = tid >> 1;
    const int akoff = (tid & 1) << 4;
    const int wwx = mT / 144, wwy = (mT / 12) % 12, wwz = mT % 12;
    const int px = arow >> 4, py = (arow >> 2) & 3, pz = arow & 3;
    const int ox = (wwx * 4 + px + 2) % 48;   // x_shifted[s] = x[s+2 mod 48]
    const int oy = (wwy * 4 + py + 2) % 48;
    const int oz = (wwz * 4 + pz + 2) % 48;
    const float* xrow = x + (size_t)(((ox * 48) + oy) * 48 + oz) * 256;

    // B-staging assignment: thread handles k-row = tid>>2, 16 n's.
    const int krow = tid >> 2;
    const int noff = (tid & 3) << 4;

    v8f acc[2][2] = {};
    for (int kb = 0; kb < 8; ++kb) {
        __syncthreads();
#pragma unroll
        for (int i = 0; i < 16; i += 2) {
            *(u32*)&As[arow * 32 + akoff + i] =
                f2bf2(xrow[kb * 32 + akoff + i], xrow[kb * 32 + akoff + i + 1]);
        }
        const u16* bsrc = wqkv + (size_t)(kb * 32 + krow) * 768 + nT * 64 + noff;
#pragma unroll
        for (int i = 0; i < 16; ++i)
            Bt[(noff + i) * 32 + krow] = bsrc[i];
        if (kb < 7) {  // prefetch next K-step tiles into near caches
            __builtin_prefetch(xrow + (kb + 1) * 32 + akoff, 0, 3);
            __builtin_prefetch(bsrc + 32 * 768, 0, 3);
        }
        __syncthreads();

        Frag a[2], b[2];
#pragma unroll
        for (int rt = 0; rt < 2; ++rt) {
            const int row = wr + rt * 16 + lm;
#pragma unroll
            for (int v = 0; v < 8; ++v) {
                const int k = ((v & 4) << 2) + half * 8 + (v & 3) * 2; // A layout
                a[rt].u[v] = *(const u32*)&As[row * 32 + k];
            }
        }
#pragma unroll
        for (int ct = 0; ct < 2; ++ct) {
            const int col = wc + ct * 16 + lm;
#pragma unroll
            for (int v = 0; v < 8; ++v) {
                const int k = ((v & 4) << 2) + half * 8 + (v & 3) * 2; // B layout
                b[ct].u[v] = *(const u32*)&Bt[col * 32 + k];
            }
        }
#pragma unroll
        for (int rt = 0; rt < 2; ++rt)
#pragma unroll
            for (int ct = 0; ct < 2; ++ct)
                acc[rt][ct] = wmma_bf16(a[rt], b[ct], acc[rt][ct]);
    }

    const float scale = 0.17677669529663687f;  // 32^-0.5 folded into Q
#pragma unroll
    for (int rt = 0; rt < 2; ++rt) {
#pragma unroll
        for (int ct = 0; ct < 2; ++ct) {
            const int n    = nT * 64 + wc + ct * 16 + lm;
            const int sec  = n >> 8;        // 0=Q 1=K 2=V (uniform per tile)
            const int f    = n & 255;
            const int head = f >> 5;
            const int d    = f & 31;
            u16* dst = (sec == 0) ? Qb : ((sec == 1) ? Kb : Vb);
            const float mul = (sec == 0) ? scale : 1.0f;
#pragma unroll
            for (int v = 0; v < 8; ++v) {
                const int pos = wr + rt * 16 + v + half * 8;           // C layout
                dst[(size_t)((mT * 8 + head) * 64 + pos) * 32 + d] =
                    f2bf(acc[rt][ct][v] * mul);
            }
        }
    }
}

// ---------------------------------------------------------------------------
// Kernel 2: windowed attention, one wave per (window, head).
// Q/K/V tiles (4KB each, contiguous) staged to LDS with async b128 copies.
// Replicates the reference's window-flatten mismatch: probabilities come from
// window (y,z,x) while V (and the output slot) belong to window (x,y,z);
// shift masks gate on y==11 (x-axis mask), z==11 (y-axis), x==11 (z-axis).
// ---------------------------------------------------------------------------
__global__ __launch_bounds__(32) void attn_win(const u16* __restrict__ Qb,
                                               const u16* __restrict__ Kb,
                                               const u16* __restrict__ Vb,
                                               u16* __restrict__ Ob) {
    __shared__ __align__(16) float S[64 * 64];  // scores (f32)
    __shared__ __align__(16) u16   P[64 * 64];  // probabilities (bf16)
    __shared__ __align__(16) u16   Qs[64 * 32];
    __shared__ __align__(16) u16   Ks[64 * 32];
    __shared__ __align__(16) u16   Vs[64 * 32];

    const int blk  = blockIdx.x;
    const int w    = blk >> 3;
    const int head = blk & 7;
    const int wx = w / 144, wy = (w / 12) % 12, wz = w % 12;
    const int wa = ((wy * 12) + wz) * 12 + wx;   // QK source window
    const int lane = threadIdx.x;
    const int half = lane >> 4;
    const int lm   = lane & 15;

    const u16* Qh = Qb + (size_t)(wa * 8 + head) * 64 * 32;
    const u16* Kh = Kb + (size_t)(wa * 8 + head) * 64 * 32;
    const u16* Vh = Vb + (size_t)(w * 8 + head) * 64 * 32;

    // ---- stage Q/K/V tiles into LDS (4KB each: 8 x 16B per lane) ----
#if HAVE_ASYNC_LDS
#pragma unroll
    for (int i = 0; i < 8; ++i) {
        const int c = (lane + i * 32) * 8;  // u16 offset of 16B chunk
        async_cp16(Qh + c, Qs + c);
        async_cp16(Kh + c, Ks + c);
        async_cp16(Vh + c, Vs + c);
    }
    async_wait0();
#else
#pragma unroll
    for (int i = 0; i < 8; ++i) {
        const int c = lane + i * 32;
        ((uint4*)Qs)[c] = ((const uint4*)Qh)[c];
        ((uint4*)Ks)[c] = ((const uint4*)Kh)[c];
        ((uint4*)Vs)[c] = ((const uint4*)Vh)[c];
    }
#endif

    // ---- S = Q @ K^T (16 WMMAs), fragments from LDS ----
    Frag aq[4], bk[4];
#pragma unroll
    for (int rt = 0; rt < 4; ++rt) {
        const int row = rt * 16 + lm;
#pragma unroll
        for (int v = 0; v < 8; ++v) {
            const int k = ((v & 4) << 2) + half * 8 + (v & 3) * 2;
            aq[rt].u[v] = *(const u32*)&Qs[row * 32 + k];
        }
    }
#pragma unroll
    for (int ct = 0; ct < 4; ++ct) {
        const int col = ct * 16 + lm;                 // key position j
#pragma unroll
        for (int v = 0; v < 8; ++v) {
            const int k = ((v & 4) << 2) + half * 8 + (v & 3) * 2;          // contraction dim d
            bk[ct].u[v] = *(const u32*)&Ks[col * 32 + k];
        }
    }
#pragma unroll
    for (int rt = 0; rt < 4; ++rt) {
#pragma unroll
        for (int ct = 0; ct < 4; ++ct) {
            v8f acc = {};
            acc = wmma_bf16(aq[rt], bk[ct], acc);
#pragma unroll
            for (int v = 0; v < 8; ++v)
                S[(rt * 16 + v + half * 8) * 64 + ct * 16 + lm] = acc[v];
        }
    }

    // ---- masked softmax, each lane owns rows (lane) and (lane+32) ----
    const bool mX = (wy == 11);
    const bool mY = (wz == 11);
    const bool mZ = (wx == 11);
#pragma unroll
    for (int rr = 0; rr < 2; ++rr) {
        const int i = lane + rr * 32;
        const int pxi = i >> 4, pyi = (i >> 2) & 3, pzi = i & 3;
        float mx = -3.0e38f;
        for (int j = 0; j < 64; ++j) {
            const int pxj = j >> 4, pyj = (j >> 2) & 3, pzj = j & 3;
            const bool masked = (mX && ((pxi >= 2) != (pxj >= 2))) ||
                                (mY && ((pyi >= 2) != (pyj >= 2))) ||
                                (mZ && ((pzi >= 2) != (pzj >= 2)));
            const float s = S[i * 64 + j];
            mx = (!masked && s > mx) ? s : mx;
        }
        float sum = 0.0f;
        for (int j = 0; j < 64; ++j) {
            const int pxj = j >> 4, pyj = (j >> 2) & 3, pzj = j & 3;
            const bool masked = (mX && ((pxi >= 2) != (pxj >= 2))) ||
                                (mY && ((pyi >= 2) != (pyj >= 2))) ||
                                (mZ && ((pzi >= 2) != (pzj >= 2)));
            const float e = masked ? 0.0f : __expf(S[i * 64 + j] - mx);
            sum += e;
            S[i * 64 + j] = e;
        }
        const float inv = 1.0f / sum;
        for (int j = 0; j < 64; j += 2)
            *(u32*)&P[i * 64 + j] =
                f2bf2(S[i * 64 + j] * inv, S[i * 64 + j + 1] * inv);
    }

    // ---- out = P @ V (16 WMMAs), write [token][head*32+d] bf16 ----
#pragma unroll
    for (int rt = 0; rt < 4; ++rt) {
        Frag ap[2];
#pragma unroll
        for (int kk = 0; kk < 2; ++kk)
#pragma unroll
            for (int v = 0; v < 8; ++v) {
                const int k = kk * 32 + ((v & 4) << 2) + half * 8 + (v & 3) * 2;
                ap[kk].u[v] = *(const u32*)&P[(rt * 16 + lm) * 64 + k];
            }
#pragma unroll
        for (int ct = 0; ct < 2; ++ct) {
            v8f acc = {};
#pragma unroll
            for (int kk = 0; kk < 2; ++kk) {
                Frag bv;
#pragma unroll
                for (int v = 0; v < 8; ++v) {
                    const int k   = kk * 32 + ((v & 4) << 2) + half * 8 + (v & 3) * 2;  // pos index
                    const int col = ct * 16 + lm;                 // d index
                    const u32 lo = Vs[k * 32 + col];
                    const u32 hi = Vs[(k + 1) * 32 + col];
                    bv.u[v] = lo | (hi << 16);
                }
                acc = wmma_bf16(ap[kk], bv, acc);
            }
#pragma unroll
            for (int v = 0; v < 8; ++v) {
                const int row = rt * 16 + v + half * 8;
                const int col = ct * 16 + lm;
                Ob[(size_t)(w * 64 + row) * 256 + head * 32 + col] = f2bf(acc[v]);
            }
        }
    }
}

// ---------------------------------------------------------------------------
// Kernel 3: output projection + bias, fused roll(+2) on the store.
// M = 110592 tokens (window-major shifted), K = 256, N = 256. fp32 out.
// A tile is a pure bf16 copy -> async global->LDS path.
// ---------------------------------------------------------------------------
__global__ __launch_bounds__(128) void out_gemm(const u16* __restrict__ Ob,
                                                const u16* __restrict__ wout,
                                                const float* __restrict__ bias,
                                                float* __restrict__ out) {
    __shared__ __align__(16) u16 As[64 * 32];
    __shared__ __align__(16) u16 Bt[64 * 32];

    const int bid  = blockIdx.x;
    const int mT   = bid >> 2;
    const int nT   = bid & 3;
    const int tid  = threadIdx.x;
    const int lane = tid & 31;
    const int half = lane >> 4;
    const int lm   = lane & 15;
    const int wave = tid >> 5;
    const int wr   = (wave >> 1) << 5;
    const int wc   = (wave & 1) << 5;

    const int arow  = tid >> 1;
    const int akoff = (tid & 1) << 4;
    const u16* asrc = Ob + (size_t)(mT * 64 + arow) * 256 + akoff;
    const int krow = tid >> 2;
    const int noff = (tid & 3) << 4;

    v8f acc[2][2] = {};
    for (int kb = 0; kb < 8; ++kb) {
        __syncthreads();
#if HAVE_ASYNC_LDS
        async_cp16(asrc + kb * 32,     &As[arow * 32 + akoff]);
        async_cp16(asrc + kb * 32 + 8, &As[arow * 32 + akoff + 8]);
#else
#pragma unroll
        for (int i = 0; i < 16; ++i)
            As[arow * 32 + akoff + i] = asrc[kb * 32 + i];
#endif
        const u16* bsrc = wout + (size_t)(kb * 32 + krow) * 256 + nT * 64 + noff;
#pragma unroll
        for (int i = 0; i < 16; ++i)
            Bt[(noff + i) * 32 + krow] = bsrc[i];
        if (kb < 7) {
            __builtin_prefetch(asrc + (kb + 1) * 32, 0, 3);
            __builtin_prefetch(bsrc + 32 * 256, 0, 3);
        }
#if HAVE_ASYNC_LDS
        async_wait0();
#endif
        __syncthreads();

        Frag a[2], b[2];
#pragma unroll
        for (int rt = 0; rt < 2; ++rt) {
            const int row = wr + rt * 16 + lm;
#pragma unroll
            for (int v = 0; v < 8; ++v) {
                const int k = ((v & 4) << 2) + half * 8 + (v & 3) * 2;
                a[rt].u[v] = *(const u32*)&As[row * 32 + k];
            }
        }
#pragma unroll
        for (int ct = 0; ct < 2; ++ct) {
            const int col = wc + ct * 16 + lm;
#pragma unroll
            for (int v = 0; v < 8; ++v) {
                const int k = ((v & 4) << 2) + half * 8 + (v & 3) * 2;
                b[ct].u[v] = *(const u32*)&Bt[col * 32 + k];
            }
        }
#pragma unroll
        for (int rt = 0; rt < 2; ++rt)
#pragma unroll
            for (int ct = 0; ct < 2; ++ct)
                acc[rt][ct] = wmma_bf16(a[rt], b[ct], acc[rt][ct]);
    }

    const int wwx = mT / 144, wwy = (mT / 12) % 12, wwz = mT % 12;
#pragma unroll
    for (int rt = 0; rt < 2; ++rt) {
#pragma unroll
        for (int ct = 0; ct < 2; ++ct) {
            const int n  = nT * 64 + wc + ct * 16 + lm;
            const float bv = bias[n];
#pragma unroll
            for (int v = 0; v < 8; ++v) {
                const int pos = wr + rt * 16 + v + half * 8;
                const int px = pos >> 4, py = (pos >> 2) & 3, pz = pos & 3;
                const int ox = (wwx * 4 + px + 2) % 48;  // roll(+2) back-shift
                const int oy = (wwy * 4 + py + 2) % 48;
                const int oz = (wwz * 4 + pz + 2) % 48;
                out[(size_t)(((ox * 48) + oy) * 48 + oz) * 256 + n] =
                    acc[rt][ct][v] + bv;
            }
        }
    }
}

// ---------------------------------------------------------------------------
extern "C" void kernel_launch(void* const* d_in, const int* in_sizes, int n_in,
                              void* d_out, int out_size, void* d_ws, size_t ws_size,
                              hipStream_t stream) {
    (void)in_sizes; (void)n_in; (void)out_size; (void)ws_size;
    const float* x     = (const float*)d_in[0];
    const float* wqkvf = (const float*)d_in[1];
    const float* woutf = (const float*)d_in[2];
    const float* bout  = (const float*)d_in[3];
    float* out = (float*)d_out;

    char* ws = (char*)d_ws;
    const size_t SZ_QKVW = 196608u * 2u;            // 393216 B
    const size_t SZ_HDR  = 524288u;                 // wqkv_bf + wout_bf (padded)
    const size_t SZ_BUF  = 28311552u * 2u;          // 56623104 B per Q/K/V/O
    u16* wqkv_bf = (u16*)(ws);
    u16* wout_bf = (u16*)(ws + SZ_QKVW);
    u16* Qb = (u16*)(ws + SZ_HDR);
    u16* Kb = (u16*)(ws + SZ_HDR + 1 * SZ_BUF);
    u16* Vb = (u16*)(ws + SZ_HDR + 2 * SZ_BUF);
    u16* Ob = (u16*)(ws + SZ_HDR + 3 * SZ_BUF);

    convert_w<<<1024, 256, 0, stream>>>(wqkvf, woutf, wqkv_bf, wout_bf);
    qkv_gemm<<<1728 * 12, 128, 0, stream>>>(x, wqkv_bf, Qb, Kb, Vb);
    attn_win<<<1728 * 8, 32, 0, stream>>>(Qb, Kb, Vb, Ob);
    out_gemm<<<1728 * 4, 128, 0, stream>>>(Ob, wout_bf, bout, out);
}
